// MultiHeadCrossAttention_17798344475176
// MI455X (gfx1250) — matrix operand order repaired and checked by reading, not compiled, not run on hardware
//
#include <hip/hip_runtime.h>
#include <hip/hip_bf16.h>
#include <stddef.h>

// ---------------------------------------------------------------------------
// Multi-head causal cross-attention for MI455X (gfx1250, wave32, WMMA).
// Shapes fixed by the reference: B=4, T=2048, C=64, H=6, D=64.
// Strategy: bf16 WMMA (f32 accum) everywhere; flash-attention fusion so the
// 400 MB score matrix never reaches HBM. Workspace layout (bf16):
//   [0]              Q   [B,H,T,D]  row-major
//   [1*BHTD]         K   [B,H,T,D]  row-major  (rows are Kt B-fragments)
//   [2*BHTD]         Vt  [B,H,D,T]  transposed (rows are V  B-fragments)
//   [3*BHTD]         A   [B,T,H*D]  attention output, pre-projection
// Total = 4*4*6*2048*64*2 bytes = 24 MB of d_ws.
// ---------------------------------------------------------------------------

typedef __attribute__((ext_vector_type(16))) __bf16 v16bf;
typedef __attribute__((ext_vector_type(8)))  __bf16 v8bf;
typedef __attribute__((ext_vector_type(8)))  float  v8f;

#define B_   4
#define T_   2048
#define C_   64
#define H_   6
#define D_   64
#define HD_  (H_ * D_)            // 384
#define BHTD ((size_t)B_ * H_ * T_ * D_)
#define NEG_INF_F (-1.0e30f)

static __device__ inline v8f wmma_bf16(v16bf a, v16bf b, v8f c) {
  return __builtin_amdgcn_wmma_f32_16x16x32_bf16(
      /*neg_a=*/false, a, /*neg_b=*/false, b,
      /*c_mod=*/(short)0, c, /*reuse_a=*/false, /*reuse_b=*/false);
}

// A-fragment (16x32, bf16): lane<16 -> row=lane, k = {0..7, 16..23};
// lane>=16 -> row=lane-16, k = {8..15, 24..31}.  src row-major, ld elems.
static __device__ inline v16bf load_a_bf16(const __bf16* src, int ld,
                                           int row0, int k0, int half, int lr) {
  const __bf16* p = src + (size_t)(row0 + lr) * ld + k0 + 8 * half;
  union { v16bf v; v8bf h[2]; } u;
  u.h[0] = *(const v8bf*)(p);
  u.h[1] = *(const v8bf*)(p + 16);
  return u.v;
}

// Same A-fragment but converting from f32 source on the fly.
static __device__ inline v16bf load_a_f32(const float* src, int ld,
                                          int row0, int k0, int half, int lr) {
  const float* p = src + (size_t)(row0 + lr) * ld + k0 + 8 * half;
  v16bf a;
#pragma unroll
  for (int e = 0; e < 8; ++e) a[e] = (__bf16)p[e];
#pragma unroll
  for (int e = 0; e < 8; ++e) a[8 + e] = (__bf16)p[16 + e];
  return a;
}

// B-fragment (32x16, bf16): lane holds column n = lane&15, rows
// k = {0..7, 16..23} (lane<16) or {8..15, 24..31} (lane>=16).  src is "n-major" (row n, contiguous k), ld elems.
static __device__ inline v16bf load_b_nmajor(const __bf16* src, int ld,
                                             int n0, int k0, int half, int lr) {
  const __bf16* p = src + (size_t)(n0 + lr) * ld + k0 + 8 * half;
  union { v16bf v; v8bf h[2]; } u;
  u.h[0] = *(const v8bf*)(p);
  u.h[1] = *(const v8bf*)(p + 16);
  return u.v;
}

static __device__ inline float rowmax16(float v) {
#pragma unroll
  for (int off = 8; off; off >>= 1) v = fmaxf(v, __shfl_xor(v, off, 16));
  return v;
}
static __device__ inline float rowsum16(float v) {
#pragma unroll
  for (int off = 8; off; off >>= 1) v += __shfl_xor(v, off, 16);
  return v;
}

// ---------------------------------------------------------------------------
// Kernel 1: QKV projections.  grid = (T/128, 3*B*H), block = 256 (8 waves).
// Each wave: one 16-row tile of [T,64] = X[b] @ W[h]  (K=64 -> 2 WMMA steps).
// ---------------------------------------------------------------------------
__global__ void __launch_bounds__(256)
qkv_proj_kernel(const float* __restrict__ x, const float* __restrict__ y,
                const float* __restrict__ Wq, const float* __restrict__ Wk,
                const float* __restrict__ Wv, __bf16* __restrict__ ws) {
  __shared__ __bf16 ldsW[D_ * C_];   // W[h] transposed: ldsW[d*64 + c], 8 KB

  const int which = blockIdx.y / (B_ * H_);     // 0=Q, 1=K, 2=V
  const int bh    = blockIdx.y % (B_ * H_);
  const int b     = bh / H_;
  const int h     = bh % H_;

  const float* W = (which == 0 ? Wq : which == 1 ? Wk : Wv) + (size_t)h * C_ * D_;
  // Stage W[h] transposed as bf16 into LDS.
  for (int i = threadIdx.x; i < C_ * D_; i += 256) {
    int c = i >> 6, d = i & 63;
    ldsW[d * C_ + c] = (__bf16)W[i];
  }
  __syncthreads();

  const int wave = threadIdx.x >> 5;
  const int lane = threadIdx.x & 31;
  const int half = lane >> 4;
  const int lr   = lane & 15;
  const int t0   = blockIdx.x * 128 + wave * 16;

  const float* src = (which == 0 ? x : y) + (size_t)b * T_ * C_;
  const v16bf a0 = load_a_f32(src, C_, t0, 0,  half, lr);
  const v16bf a1 = load_a_f32(src, C_, t0, 32, half, lr);

  __bf16* Qb = ws;
  __bf16* Kb = ws + BHTD;
  __bf16* Vt = ws + 2 * BHTD;

#pragma unroll
  for (int nt = 0; nt < 4; ++nt) {
    v8f acc = {};
    acc = wmma_bf16(a0, load_b_nmajor(ldsW, C_, nt * 16, 0,  half, lr), acc);
    acc = wmma_bf16(a1, load_b_nmajor(ldsW, C_, nt * 16, 32, half, lr), acc);
    if (which < 2) {                       // Q / K: row-major [bh][t][d]
      __bf16* dst = (which == 0 ? Qb : Kb) + (size_t)bh * T_ * D_;
#pragma unroll
      for (int r = 0; r < 8; ++r)
        dst[(size_t)(t0 + r + 8 * half) * D_ + nt * 16 + lr] = (__bf16)acc[r];
    } else {                               // V: transposed [bh][d][t]
      __bf16* dst = Vt + (size_t)bh * D_ * T_;
#pragma unroll
      for (int r = 0; r < 8; ++r)
        dst[(size_t)(nt * 16 + lr) * T_ + t0 + r + 8 * half] = (__bf16)acc[r];
    }
  }
}

// ---------------------------------------------------------------------------
// Kernel 2: flash attention.  grid = (T/128, B*H), block = 256 (8 waves).
// One wave owns a 16-row query tile; streams keys 32 at a time.
// Causal load balance: query tile index = wave*16 + blockIdx.x, so every
// block's 8 waves sample the whole 0..127 tile range -> per-block work is
// nearly uniform instead of 16x skewed.
// ---------------------------------------------------------------------------
__global__ void __launch_bounds__(256)
flash_attn_kernel(const __bf16* __restrict__ ws, __bf16* __restrict__ Abf) {
  __shared__ __bf16 pslab[8][16 * 32];     // wave-private P re-layout, 8 KB

  const int bh = blockIdx.y;
  const int b  = bh / H_;
  const int h  = bh % H_;

  const int wave = threadIdx.x >> 5;
  const int lane = threadIdx.x & 31;
  const int half = lane >> 4;
  const int lr   = lane & 15;
  const int q0   = (wave * (T_ / 128) + blockIdx.x) * 16;   // balanced map

  const __bf16* Qb = ws            + (size_t)bh * T_ * D_;
  const __bf16* Kb = ws + BHTD     + (size_t)bh * T_ * D_;
  const __bf16* Vt = ws + 2 * BHTD + (size_t)bh * D_ * T_;

  const v16bf qa0 = load_a_bf16(Qb, D_, q0, 0,  half, lr);
  const v16bf qa1 = load_a_bf16(Qb, D_, q0, 32, half, lr);

  float m_i[8], l_i[8], alpha[8];
  v8f acc[4] = {};
#pragma unroll
  for (int r = 0; r < 8; ++r) { m_i[r] = -3.0e38f; l_i[r] = 0.0f; }

  const float scale = 0.125f;              // D^-0.5
  const int   sEnd  = q0 + 15;             // causal: keys <= query row

  for (int s0 = 0; s0 <= sEnd; s0 += 32) {
    // Prefetch next iteration's K/V fragment cachelines (global_prefetch_b8).
    if (s0 + 32 <= sEnd) {
      __builtin_prefetch(Kb + (size_t)(s0 + 32 + lr) * D_ + 16 * half, 0, 3);
      __builtin_prefetch(Vt + (size_t)lr * T_ + s0 + 32 + 16 * half, 0, 3);
    }

    // Scores for 32 keys: two 16x16 tiles, K-dim = 64 -> 2 WMMAs each.
    v8f sc0 = {}, sc1 = {};
    sc0 = wmma_bf16(qa0, load_b_nmajor(Kb, D_, s0,      0,  half, lr), sc0);
    sc0 = wmma_bf16(qa1, load_b_nmajor(Kb, D_, s0,      32, half, lr), sc0);
    sc1 = wmma_bf16(qa0, load_b_nmajor(Kb, D_, s0 + 16, 0,  half, lr), sc1);
    sc1 = wmma_bf16(qa1, load_b_nmajor(Kb, D_, s0 + 16, 32, half, lr), sc1);

    // Mask + online softmax (row M = r + 8*half, key N = lane&15).
#pragma unroll
    for (int r = 0; r < 8; ++r) {
      const int qm = q0 + r + 8 * half;
      float e0 = (s0 + lr      <= qm) ? sc0[r] * scale : NEG_INF_F;
      float e1 = (s0 + 16 + lr <= qm) ? sc1[r] * scale : NEG_INF_F;
      float mx = rowmax16(fmaxf(e0, e1));
      float mn = fmaxf(m_i[r], mx);
      float al = __expf(m_i[r] - mn);
      float p0 = __expf(e0 - mn);
      float p1 = __expf(e1 - mn);
      l_i[r] = l_i[r] * al + rowsum16(p0 + p1);
      m_i[r] = mn;
      alpha[r] = al;
      // Stage P (16x32, bf16) row-major into the wave's LDS slab.
      pslab[wave][(r + 8 * half) * 32 + lr]      = (__bf16)p0;
      pslab[wave][(r + 8 * half) * 32 + 16 + lr] = (__bf16)p1;
    }

    // Rescale running output, then O += P(16x32) @ V(32x64).
#pragma unroll
    for (int nt = 0; nt < 4; ++nt)
#pragma unroll
      for (int r = 0; r < 8; ++r) acc[nt][r] *= alpha[r];

    const v16bf pa = load_a_bf16(&pslab[wave][0], 32, 0, 0, half, lr);
#pragma unroll
    for (int nt = 0; nt < 4; ++nt)
      acc[nt] = wmma_bf16(pa, load_b_nmajor(Vt, T_, nt * 16, s0, half, lr),
                          acc[nt]);
  }

  // Normalize and write head output into concat layout [B, T, H*D] (bf16).
#pragma unroll
  for (int r = 0; r < 8; ++r) alpha[r] = 1.0f / l_i[r];
#pragma unroll
  for (int nt = 0; nt < 4; ++nt)
#pragma unroll
    for (int r = 0; r < 8; ++r)
      Abf[((size_t)b * T_ + q0 + r + 8 * half) * HD_ + h * D_ + nt * 16 + lr] =
          (__bf16)(acc[nt][r] * alpha[r]);
}

// ---------------------------------------------------------------------------
// Kernel 3: output projection  out = A[BT,384] @ W_proj[384,64] + b_proj.
// grid = (B*T/128), block = 256 (8 waves).  W_proj lives transposed in LDS.
// ---------------------------------------------------------------------------
__global__ void __launch_bounds__(256)
out_proj_kernel(const __bf16* __restrict__ Abf,
                const float* __restrict__ W_proj,
                const float* __restrict__ b_proj,
                float* __restrict__ out) {
  __shared__ __bf16 ldsW[C_ * HD_];        // Wt[n*384 + k], 48 KB of 320 KB

  for (int i = threadIdx.x; i < HD_ * C_; i += 256) {
    int k = i / C_, n = i % C_;
    ldsW[n * HD_ + k] = (__bf16)W_proj[i];
  }
  __syncthreads();

  const int wave = threadIdx.x >> 5;
  const int lane = threadIdx.x & 31;
  const int half = lane >> 4;
  const int lr   = lane & 15;
  const int t0   = blockIdx.x * 128 + wave * 16;   // row in [B*T, 384]

  v8f acc[4] = {};
#pragma unroll
  for (int ks = 0; ks < HD_ / 32; ++ks) {          // 12 K-steps
    const v16bf a = load_a_bf16(Abf, HD_, t0, 32 * ks, half, lr);
#pragma unroll
    for (int nt = 0; nt < 4; ++nt)
      acc[nt] = wmma_bf16(a, load_b_nmajor(ldsW, HD_, nt * 16, 32 * ks,
                                           half, lr), acc[nt]);
  }

#pragma unroll
  for (int nt = 0; nt < 4; ++nt) {
    const float bias = b_proj[nt * 16 + lr];
#pragma unroll
    for (int r = 0; r < 8; ++r)
      out[(size_t)(t0 + r + 8 * half) * C_ + nt * 16 + lr] = acc[nt][r] + bias;
  }
}

// ---------------------------------------------------------------------------
extern "C" void kernel_launch(void* const* d_in, const int* in_sizes, int n_in,
                              void* d_out, int out_size, void* d_ws,
                              size_t ws_size, hipStream_t stream) {
  const float* x      = (const float*)d_in[0];
  const float* y      = (const float*)d_in[1];
  const float* Wq     = (const float*)d_in[2];
  const float* Wk     = (const float*)d_in[3];
  const float* Wv     = (const float*)d_in[4];
  const float* W_proj = (const float*)d_in[5];
  const float* b_proj = (const float*)d_in[6];
  float* out = (float*)d_out;

  __bf16* ws  = (__bf16*)d_ws;             // needs 4*BHTD*2 = 24 MB
  __bf16* Abf = ws + 3 * BHTD;

  qkv_proj_kernel<<<dim3(T_ / 128, 3 * B_ * H_), 256, 0, stream>>>(
      x, y, Wq, Wk, Wv, ws);
  flash_attn_kernel<<<dim3(T_ / 128, B_ * H_), 256, 0, stream>>>(ws, Abf);
  out_proj_kernel<<<dim3((B_ * T_) / 128), 256, 0, stream>>>(
      Abf, W_proj, b_proj, out);
}
